// DiffPoolGCN_30855045055189
// MI455X (gfx1250) — matrix operand order repaired and checked by reading, not compiled, not run on hardware
//
#include <hip/hip_runtime.h>

typedef __attribute__((ext_vector_type(16))) _Float16 v16h;
typedef __attribute__((ext_vector_type(8)))  float    v8f;

#define FEAT 128

// ---------------- degree / norm ----------------
__global__ void k_init_deg(float* __restrict__ deg, int n) {
    int i = blockIdx.x * blockDim.x + threadIdx.x;
    if (i < n) deg[i] = 1.0f;   // self loop contributes 1
}

__global__ void k_count_deg(const long long* __restrict__ dst, long long e,
                            float* __restrict__ deg) {
    long long i = (long long)blockIdx.x * blockDim.x + threadIdx.x;
    if (i < e) unsafeAtomicAdd(&deg[dst[i]], 1.0f);
}

__global__ void k_rsqrt(const float* __restrict__ deg, float* __restrict__ dinv, int n) {
    int i = blockIdx.x * blockDim.x + threadIdx.x;
    if (i < n) dinv[i] = rsqrtf(deg[i]);
}

// ---------------- W (f32 row-major) -> Wt (f16 col-major) ----------------
__global__ void k_convert_w(const float* __restrict__ W, _Float16* __restrict__ Wt) {
    int t = blockIdx.x * blockDim.x + threadIdx.x;   // 128*128 threads
    int col = t >> 7, k = t & 127;
    Wt[(size_t)col * FEAT + k] = (_Float16)W[(size_t)k * FEAT + col];
}

// ---------------- WMMA GEMM: H[N,128] = X[N,128] * W[128,128] ----------------
// block = 256 threads (8 waves). Each wave: one 16x16 output tile.
// wave w covers columns [16w, 16w+16); block covers rows [16*blockIdx.x, +16).
__global__ void __launch_bounds__(256) k_gemm(const float* __restrict__ X,
                                              const _Float16* __restrict__ Wt,
                                              float* __restrict__ H, int n) {
    const int lane = threadIdx.x & 31;
    const int wave = threadIdx.x >> 5;
    const int m0   = blockIdx.x * 16;
    const int n0   = wave * 16;
    const int ml   = lane & 15;
    const int hi   = lane >> 4;          // 0 or 1

    int mrow = m0 + ml;
    if (mrow >= n) mrow = n - 1;         // clamp (EXEC must stay all-1s for WMMA)
    const float* xrow = X + (size_t)mrow * FEAT;

    v8f acc = {};
#pragma unroll
    for (int ks = 0; ks < 4; ++ks) {
        // A fragment: lane holds row m0+ml; elems 0..15 -> K=kb..kb+15,
        // kb = 32*ks + 16*hi  (contiguous order, matches B)
        const int kb = ks * 32 + hi * 16;
        v16h a;
#pragma unroll
        for (int e = 0; e < 8; ++e) {
            a[e]     = (_Float16)xrow[kb + e];
            a[e + 8] = (_Float16)xrow[kb + 8 + e];
        }
        // B fragment: lane holds column n0+ml; elems 0..15 -> K = 32*ks + 16*hi + e.
        // Wt is col-major f16, so this is one contiguous 32B load.
        const v16h b = *(const v16h*)(Wt + (size_t)(n0 + ml) * FEAT + ks * 32 + hi * 16);

        acc = __builtin_amdgcn_wmma_f32_16x16x32_f16(
            /*neg_a=*/false, a, /*neg_b=*/false, b,
            /*c_mod=*/(short)0, acc, /*reuse_a=*/false, /*reuse_b=*/false);
    }
    // C/D layout: VGPR r, lane l -> row r + 8*hi, col n0 + ml
#pragma unroll
    for (int r = 0; r < 8; ++r) {
        int row = m0 + r + 8 * hi;
        if (row < n) H[(size_t)row * FEAT + n0 + ml] = acc[r];
    }
}

// ---------------- self-loop init: A = H * dinv^2 ----------------
__global__ void k_selfloop(const float* __restrict__ H, const float* __restrict__ dinv,
                           float* __restrict__ A, int n) {
    int t = blockIdx.x * blockDim.x + threadIdx.x;   // n*32 threads, float4 each
    if (t >= n * 32) return;
    int i = t >> 5, c = (t & 31) * 4;
    float d = dinv[i], s = d * d;
    float4 v = *(const float4*)(H + (size_t)i * FEAT + c);
    float4 o = {v.x * s, v.y * s, v.z * s, v.w * s};
    *(float4*)(A + (size_t)i * FEAT + c) = o;
}

// ---------------- edge scatter: A[dst] += H[src] * dinv[src]*dinv[dst] ----------------
// one wave per edge, 4 channels per lane
__global__ void __launch_bounds__(256) k_scatter(const long long* __restrict__ src,
                                                 const long long* __restrict__ dst,
                                                 long long e,
                                                 const float* __restrict__ H,
                                                 const float* __restrict__ dinv,
                                                 float* __restrict__ A) {
    long long eid = (long long)blockIdx.x * (blockDim.x >> 5) + (threadIdx.x >> 5);
    if (eid >= e) return;
    const int lane = threadIdx.x & 31;
    const long long s = src[eid];
    const long long d = dst[eid];
    const float norm = dinv[s] * dinv[d];
    const int c = lane * 4;
    float4 v = *(const float4*)(H + (size_t)s * FEAT + c);
    float* ap = A + (size_t)d * FEAT + c;
    unsafeAtomicAdd(ap + 0, v.x * norm);
    unsafeAtomicAdd(ap + 1, v.y * norm);
    unsafeAtomicAdd(ap + 2, v.z * norm);
    unsafeAtomicAdd(ap + 3, v.w * norm);
}

// ---------------- epilogue: out = relu(A + b) ----------------
__global__ void k_epilogue(const float* __restrict__ A, const float* __restrict__ bias,
                           float* __restrict__ out, int n) {
    int t = blockIdx.x * blockDim.x + threadIdx.x;
    if (t >= n * 32) return;
    int i = t >> 5, c = (t & 31) * 4;
    float4 v = *(const float4*)(A + (size_t)i * FEAT + c);
    float4 b = *(const float4*)(bias + c);
    float4 o = {fmaxf(v.x + b.x, 0.0f), fmaxf(v.y + b.y, 0.0f),
                fmaxf(v.z + b.z, 0.0f), fmaxf(v.w + b.w, 0.0f)};
    *(float4*)(out + (size_t)i * FEAT + c) = o;
}

extern "C" void kernel_launch(void* const* d_in, const int* in_sizes, int n_in,
                              void* d_out, int out_size, void* d_ws, size_t ws_size,
                              hipStream_t stream) {
    const float*     x  = (const float*)d_in[0];
    const long long* ei = (const long long*)d_in[1];   // int64 [2, E]
    const float* W[3] = {(const float*)d_in[2], (const float*)d_in[4], (const float*)d_in[6]};
    const float* b[3] = {(const float*)d_in[3], (const float*)d_in[5], (const float*)d_in[7]};

    const int       n = in_sizes[0] / FEAT;    // 100000
    const long long e = in_sizes[1] / 2;       // 1600000
    const long long* src = ei;
    const long long* dst = ei + e;

    // workspace carve-up
    float*     H    = (float*)d_ws;                         // [N,128]
    float*     A    = H + (size_t)n * FEAT;                 // [N,128]
    float*     deg  = A + (size_t)n * FEAT;                 // [N]
    float*     dinv = deg + n;                              // [N]
    _Float16*  Wt   = (_Float16*)(dinv + n);                // [128,128] f16 col-major

    const int TB = 256;
    dim3 blk(TB);

    // degrees + normalization
    k_init_deg<<<dim3((n + TB - 1) / TB), blk, 0, stream>>>(deg, n);
    k_count_deg<<<dim3((unsigned)((e + TB - 1) / TB)), blk, 0, stream>>>(dst, e, deg);
    k_rsqrt<<<dim3((n + TB - 1) / TB), blk, 0, stream>>>(deg, dinv, n);

    const unsigned gemm_grid  = (unsigned)((n + 15) / 16);
    const unsigned elem_grid  = (unsigned)(((size_t)n * 32 + TB - 1) / TB);
    const unsigned edge_grid  = (unsigned)((e + 7) / 8);   // 8 edges (waves) per block

    const float* feat = x;
    for (int l = 0; l < 3; ++l) {
        k_convert_w<<<dim3(64), blk, 0, stream>>>(W[l], Wt);
        k_gemm<<<dim3(gemm_grid), blk, 0, stream>>>(feat, Wt, H, n);
        k_selfloop<<<dim3(elem_grid), blk, 0, stream>>>(H, dinv, A, n);
        k_scatter<<<dim3(edge_grid), blk, 0, stream>>>(src, dst, e, H, dinv, A);
        float* outp = (l == 2) ? (float*)d_out : A;        // in-place relu for l<2
        k_epilogue<<<dim3(elem_grid), blk, 0, stream>>>(A, b[l], outp, n);
        feat = A;                                          // next layer reads features from A
    }
}
